// LipCNNConv_49572512530826
// MI455X (gfx1250) — matrix operand order repaired and checked by reading, not compiled, not run on hardware
//
#include <hip/hip_runtime.h>
#include <hip/hip_bf16.h>

// ---------------------------------------------------------------------------
// Types for WMMA
// ---------------------------------------------------------------------------
typedef _Float16 v16h __attribute__((ext_vector_type(16)));
typedef _Float16 v8h  __attribute__((ext_vector_type(8)));
typedef float    v8f  __attribute__((ext_vector_type(8)));

#define COUT 128
#define CIN  128
#define HH   56
#define WW   56
#define BB   64
#define KTOT 1152          // CIN*9
#define KTILE 288          // lcm(32,9): 9 WMMA chunks per K-iteration
#define NKIT (KTOT / KTILE)
#define MT   64            // pixels per block (four 16-wide M tiles)
#define ASTRIDE 296        // padded LDS row stride (halves)
#define CHSTEP (32 * HH * WW)        // x16 offset step per K-iteration (32 ci)
#define NPIX (BB * CIN * HH * WW)    // total x elements

// ---------------------------------------------------------------------------
// Small linear-algebra utility kernels (preprocessing; negligible FLOPs)
// ---------------------------------------------------------------------------
__global__ void k_fill(float* p, int n, float v) {
  int i = blockIdx.x * blockDim.x + threadIdx.x;
  if (i < n) p[i] = v;
}

__global__ void k_cvt(_Float16* x16, const float* x, int n) {
  int i = blockIdx.x * blockDim.x + threadIdx.x;
  if (i < n) x16[i] = (_Float16)x[i];
}

__global__ void k_axpy(float* y, const float* x, int n, float a) {
  int i = blockIdx.x * blockDim.x + threadIdx.x;
  if (i < n) y[i] += a * x[i];
}

__global__ void k_copy_block(float* dst, int ldd, const float* src, int lds,
                             int rows, int cols, float alpha, float beta) {
  int idx = blockIdx.x * blockDim.x + threadIdx.x;
  if (idx >= rows * cols) return;
  int i = idx / cols, j = idx - i * cols;
  float prev = (beta != 0.f) ? dst[i * ldd + j] : 0.f;
  dst[i * ldd + j] = alpha * src[i * lds + j] + beta * prev;
}

__global__ void k_add_eye(float* M, int n, int ld, float v) {
  int i = blockIdx.x * blockDim.x + threadIdx.x;
  if (i < n) M[i * ld + i] += v;
}

__global__ void k_set_diag(float* p, int ld, const float* vec, float v, int n) {
  int i = blockIdx.x * blockDim.x + threadIdx.x;
  if (i < n) p[i * ld + i] = vec ? vec[i] : v;
}

__global__ void k_symmetrize(float* M, int n, int ld) {
  int idx = blockIdx.x * blockDim.x + threadIdx.x;
  if (idx >= n * n) return;
  int i = idx / n, j = idx - i * n;
  if (i < j) {
    float v = 0.5f * (M[i * ld + j] + M[j * ld + i]);
    M[i * ld + j] = v;
    M[j * ld + i] = v;
  }
}

// C = alpha*op(A)op(B) + beta*C   (row-major, naive; matrices <= 640)
__global__ void k_gemm(float* C, const float* A, const float* B,
                       int M, int N, int K, int lda, int ldb, int ldc,
                       int tA, int tB, float alpha, float beta) {
  int idx = blockIdx.x * blockDim.x + threadIdx.x;
  if (idx >= M * N) return;
  int m = idx / N, n = idx - m * N;
  float s = 0.f;
  for (int k = 0; k < K; ++k) {
    float av = tA ? A[k * lda + m] : A[m * lda + k];
    float bv = tB ? B[n * ldb + k] : B[k * ldb + n];
    s += av * bv;
  }
  float prev = (beta != 0.f) ? C[m * ldc + n] : 0.f;
  C[m * ldc + n] = alpha * s + beta * prev;
}

// Gauss-Jordan inverse, single block (n <= 256), aug scratch n x 2n in global.
__global__ void k_inverse(float* out, const float* src, int n, float* aug) {
  const int tid = threadIdx.x;
  const int nt  = blockDim.x;
  const int n2  = 2 * n;
  __shared__ float s_col[256];
  __shared__ float s_piv;
  for (int e = tid; e < n * n2; e += nt) {
    int i = e / n2, j = e - i * n2;
    aug[e] = (j < n) ? src[i * n + j] : ((j - n == i) ? 1.f : 0.f);
  }
  __syncthreads();
  for (int p = 0; p < n; ++p) {
    if (tid == 0) s_piv = 1.f / aug[p * n2 + p];
    __syncthreads();
    float pv = s_piv;
    for (int j = tid; j < n2; j += nt) aug[p * n2 + j] *= pv;
    for (int i = tid; i < n; i += nt) s_col[i] = aug[i * n2 + p];
    __syncthreads();
    for (int e = tid; e < n * n2; e += nt) {
      int i = e / n2, j = e - i * n2;
      if (i != p) aug[e] -= s_col[i] * aug[p * n2 + j];
    }
    __syncthreads();
  }
  for (int e = tid; e < n * n; e += nt) {
    int i = e / n, j = e - i * n;
    out[e] = aug[i * n2 + n + j];
  }
}

// In-place lower Cholesky, single block (n <= 640); zeroes strict upper.
__global__ void k_cholesky(float* A, int n, int ld) {
  const int tid = threadIdx.x;
  const int nt  = blockDim.x;
  __shared__ float s_d;
  for (int k = 0; k < n; ++k) {
    if (tid == 0) {
      float d = sqrtf(A[k * ld + k]);
      A[k * ld + k] = d;
      s_d = d;
    }
    __syncthreads();
    float d = s_d;
    for (int i = k + 1 + tid; i < n; i += nt) A[i * ld + k] /= d;
    __syncthreads();
    int r = n - 1 - k;
    for (int e = tid; e < r * r; e += nt) {
      int i = k + 1 + e / r;
      int j = k + 1 + e % r;
      if (j <= i) A[i * ld + j] -= A[i * ld + k] * A[j * ld + k];
    }
    __syncthreads();
  }
  for (int e = tid; e < n * n; e += nt) {
    int i = e / n, j = e - i * n;
    if (j > i) A[i * ld + j] = 0.f;
  }
}

// Forward substitution: solve L * Y = B in place (one thread per RHS column).
__global__ void k_trsm(const float* L, int ldl, float* Bio, int ldb, int n, int r) {
  int t = threadIdx.x;
  if (t >= r) return;
  for (int i = 0; i < n; ++i) {
    float s = Bio[i * ldb + t];
    for (int j = 0; j < i; ++j) s -= L[i * ldl + j] * Bio[j * ldb + t];
    Bio[i * ldb + t] = s / L[i * ldl + i];
  }
}

__global__ void k_trace(float* tr, const float* G, int n, int ld) {
  if (threadIdx.x == 0 && blockIdx.x == 0) {
    float s = 0.f;
    for (int i = 0; i < n; ++i) s += G[i * ld + i];
    *tr = s;
  }
}

__global__ void k_make_sym(float* Sym, const float* G, const float* trp, int n) {
  int idx = blockIdx.x * blockDim.x + threadIdx.x;
  if (idx >= n * n) return;
  int i = idx / n, j = idx - i * n;
  float sc = 1.f / (1.f + *trp);
  Sym[idx] = ((i == j) ? 1.f : 0.f) - G[idx] * sc;
}

__global__ void k_skewadd(float* Ac, const float* U0, int n) {
  int idx = blockIdx.x * blockDim.x + threadIdx.x;
  if (idx >= n * n) return;
  int i = idx / n, j = idx - i * n;
  Ac[idx] += U0[i * n + j] - U0[j * n + i];
}

__global__ void k_divrow(float* M, const float* q, int rows, int cols) {
  int idx = blockIdx.x * blockDim.x + threadIdx.x;
  if (idx >= rows * cols) return;
  int i = idx / cols;
  M[idx] /= q[i];
}

// Pack final 384x384 'mat' into f16 weights, fragment-friendly [co][k] layout.
// Kk[co,ci,kh,kw] = mat[(2-kh)*128+co, (2-kw)*128+ci],  k = ci*9+kh*3+kw
__global__ void k_pack(_Float16* Kh, const float* mat) {
  int idx = blockIdx.x * blockDim.x + threadIdx.x;
  if (idx >= COUT * KTOT) return;
  int co = idx / KTOT, k = idx - co * KTOT;
  int ci = k / 9, r = k - ci * 9;
  int kh = r / 3, kw = r - kh * 3;
  Kh[idx] = (_Float16)mat[((2 - kh) * COUT + co) * 384 + ((2 - kw) * CIN + ci)];
}

// ---------------------------------------------------------------------------
// Implicit-GEMM conv via V_WMMA_F32_16X16X32_F16.
// Block: 64 pixels (one full h row) x 128 couts; 8 waves, each wave owns
// four 16x16 f32 accumulators -> 36 v_wmma per wave per K-iteration.
// K steps by 288 (= lcm(32,9)); each thread gathers 72 contiguous K-values
// of one pixel (8 whole (ci,kh,kw) 9-groups) from the pre-converted f16
// image (L2-resident) and stores them as 9 x ds_store_b128.
// B fragments are direct 32B global loads from L2-resident packed weights.
// ---------------------------------------------------------------------------
__global__ __launch_bounds__(256) void k_conv_wmma(
    float* __restrict__ out, const _Float16* __restrict__ x16,
    const _Float16* __restrict__ Kh, const float* __restrict__ bias)
{
  __shared__ __align__(32) _Float16 Atile[MT * ASTRIDE];

  const int n   = blockIdx.z;
  const int h   = blockIdx.y;
  const int tid = threadIdx.x;
  const int lane = tid & 31;
  const int wave = tid >> 5;
  const int row0 = lane & 15;         // M row within a tile / D column
  const int hi   = (lane >> 4) & 1;   // half-wave selector
  const int co   = wave * 16 + row0;  // this lane's output channel

  // Gather geometry: thread owns pixel m = tid/4, K-range [kk0, kk0+72).
  const int m    = tid >> 2;          // 0..63 (pixel w index; w0 == 0)
  const int kk0  = (tid & 3) * 72;    // 8 complete 9-groups
  const int ciT  = (tid & 3) * 8;     // first ci offset within 32-ci chunk
  const int iw0  = m;
  const bool vh[3] = { h > 0, true, h < HH - 1 };
  const bool vw[3] = { (iw0 >= 1) && (iw0 <= WW), iw0 < WW, iw0 + 1 < WW };

  v8f acc[4];
#pragma unroll
  for (int mt = 0; mt < 4; ++mt)
    acc[mt] = (v8f){0.f, 0.f, 0.f, 0.f, 0.f, 0.f, 0.f, 0.f};

  const _Float16* xblk  = x16 + (size_t)n * CIN * HH * WW;
  const _Float16* Bbase = Kh + (size_t)co * KTOT + hi * 16;
  const _Float16* Arow  = &Atile[row0 * ASTRIDE + hi * 16];
  _Float16*       dst   = &Atile[m * ASTRIDE + kk0];

  for (int t = 0; t < NKIT; ++t) {
    __syncthreads();
    // Gather 72 halves (8 groups x 3 kh x 3 kw) into registers
    v8h lbuf[9];
#pragma unroll
    for (int g = 0; g < 8; ++g) {
      const _Float16* cbase = xblk + (ciT + g) * (HH * WW);
#pragma unroll
      for (int kh = 0; kh < 3; ++kh) {
        const _Float16* src = cbase + (h + kh - 1) * WW + iw0 - 1;
#pragma unroll
        for (int kw = 0; kw < 3; ++kw) {
          int e = g * 9 + kh * 3 + kw;
          _Float16 v = (_Float16)0.f;
          if (vh[kh] && vw[kw]) v = src[kw];
          lbuf[e >> 3][e & 7] = v;
        }
      }
    }
    // 9 x ds_store_b128 (16B-aligned by construction)
#pragma unroll
    for (int j = 0; j < 9; ++j) *(v8h*)(dst + j * 8) = lbuf[j];
    // Speculative prefetch of next ci-block (global_prefetch_b8)
    if (t < NKIT - 1)
      __builtin_prefetch(xblk + CHSTEP + ciT * (HH * WW) + h * WW + iw0, 0, 0);
    __syncthreads();

    const _Float16* Bp = Bbase + t * KTILE;
#pragma unroll
    for (int s = 0; s < 9; ++s) {
      // B fragment: 16 contiguous halves of this co's K-stream (32B load)
      v16h bv = *(const v16h*)(Bp + s * 32);
#pragma unroll
      for (int mt = 0; mt < 4; ++mt) {
        const _Float16* ap = Arow + mt * 16 * ASTRIDE + s * 32;
        v8h a0 = *(const v8h*)(ap);
        v8h a1 = *(const v8h*)(ap + 8);
        v16h av = __builtin_shufflevector(a0, a1, 0, 1, 2, 3, 4, 5, 6, 7,
                                                  8, 9, 10, 11, 12, 13, 14, 15);
        acc[mt] = __builtin_amdgcn_wmma_f32_16x16x32_f16(
            false, av, false, bv, (short)0, acc[mt], false, false);
      }
    }
    xblk += CHSTEP;
  }

  const float bvl = bias[co];
  float* op = out + (((size_t)n * COUT + co) * HH + h) * WW;
#pragma unroll
  for (int mt = 0; mt < 4; ++mt) {
#pragma unroll
    for (int r = 0; r < 8; ++r) {
      int w = mt * 16 + r + hi * 8;   // D layout: VGPR r -> M = r + 8*hi
      if (w < WW) op[w] = acc[mt][r] + bvl;
    }
  }
}

// ---------------------------------------------------------------------------
// Host orchestration
// ---------------------------------------------------------------------------
extern "C" void kernel_launch(void* const* d_in, const int* in_sizes, int n_in,
                              void* d_out, int out_size, void* d_ws, size_t ws_size,
                              hipStream_t stream) {
  const float* x   = (const float*)d_in[0];
  const float* Q   = (const float*)d_in[1];
  const float* wt  = (const float*)d_in[2];   // (512,128)
  const float* bia = (const float*)d_in[3];
  const float* A12 = (const float*)d_in[4];   // (256,256)
  const float* B1  = (const float*)d_in[5];   // (256,128)
  const float* H1  = (const float*)d_in[6];   // (256,256)
  const float* H2  = (const float*)d_in[7];   // (256,256)
  const float* qv  = (const float*)d_in[8];   // (128,)
  float* out = (float*)d_out;
  float* W = (float*)d_ws;

  size_t off = 0;
  auto alloc = [&](size_t nf) { size_t o = off; off += nf; return W + o; };
  float* QINV = alloc(128 * 128);
  float* BQ   = alloc(256 * 128);
  float* XT11 = alloc(256 * 256);
  float* XT22 = alloc(256 * 256);
  float* T2   = alloc(256 * 256);
  float* CR   = alloc(256 * 256);   // cross
  float* INA  = alloc(256 * 256);
  float* INV2 = alloc(256 * 256);   // inner
  float* M1   = alloc(256 * 256);   // also T1
  float* TMPA = alloc(512 * 512);
  float* TMPB = alloc(512 * 512);
  float* P    = alloc(512 * 512);
  float* AM   = alloc(512 * 512);
  float* BM   = alloc(512 * 128);
  float* PB   = alloc(512 * 128);
  float* F    = alloc(640 * 640);
  float* Y    = alloc(256 * 128);
  float* G    = alloc(128 * 128);
  float* SYM  = alloc(128 * 128);
  float* AC   = alloc(128 * 128);
  float* ACT  = alloc(128 * 128);
  float* IIPA = alloc(128 * 128);
  float* IMA  = alloc(128 * 128);
  float* UCAL = alloc(384 * 128);
  float* Z2T  = alloc(128 * 384);
  float* C2D  = alloc(128 * 384);
  float* MAT  = alloc(384 * 384);
  float* TR   = alloc(8);
  float* AUG  = alloc(256 * 512);
  _Float16* KH  = (_Float16*)alloc(COUT * KTOT / 2);  // f16 packed weights
  _Float16* X16 = (_Float16*)alloc(NPIX / 2);         // f16 image (L2-resident)

  auto fill = [&](float* p, int n, float v) {
    k_fill<<<(n + 255) / 256, 256, 0, stream>>>(p, n, v);
  };
  auto axpy = [&](float* y, const float* xs, int n, float a) {
    k_axpy<<<(n + 255) / 256, 256, 0, stream>>>(y, xs, n, a);
  };
  auto cpb = [&](float* d, int ldd, const float* s, int lds, int r, int c,
                 float al, float be) {
    k_copy_block<<<(r * c + 255) / 256, 256, 0, stream>>>(d, ldd, s, lds, r, c, al, be);
  };
  auto gemm = [&](float* C, const float* Ag, const float* Bg, int M, int N, int K,
                  int lda, int ldb, int ldc, int tA, int tB, float al, float be) {
    k_gemm<<<(M * N + 255) / 256, 256, 0, stream>>>(C, Ag, Bg, M, N, K, lda, ldb,
                                                    ldc, tA, tB, al, be);
  };
  auto inv = [&](float* o, const float* s, int n) {
    k_inverse<<<1, 1024, 0, stream>>>(o, s, n, AUG);
  };
  const float EPSV = 1e-6f;

  // 0. Pre-convert image to f16 (overlaps with preprocessing chain)
  k_cvt<<<(NPIX + 255) / 256, 256, 0, stream>>>(X16, x, NPIX);
  // 1. Qinv
  inv(QINV, Q, 128);
  // 2. BQ = B1 @ Qinv ; Xt11 = BQ @ B1^T (symmetrized)
  gemm(BQ, B1, QINV, 256, 128, 128, 128, 128, 128, 0, 0, 1.f, 0.f);
  gemm(XT11, BQ, B1, 256, 256, 128, 128, 128, 256, 0, 1, 1.f, 0.f);
  k_symmetrize<<<(256 * 256 + 255) / 256, 256, 0, stream>>>(XT11, 256, 256);
  // 3. Xt22 = embed(Qinv) bottom-right
  fill(XT22, 256 * 256, 0.f);
  cpb(XT22 + 128 * 256 + 128, 256, QINV, 128, 128, 128, 1.f, 0.f);
  // 4. M2 -> T2 buf: H2^T H2 + Xt22 + eps I
  gemm(T2, H2, H2, 256, 256, 256, 256, 256, 256, 1, 0, 1.f, 0.f);
  axpy(T2, XT22, 256 * 256, 1.f);
  k_add_eye<<<1, 256, 0, stream>>>(T2, 256, 256, EPSV);
  // 5. T2 = M2 + A22 M2 A22^T (TL += BR)
  cpb(T2, 256, T2 + 128 * 256 + 128, 256, 128, 128, 1.f, 1.f);
  // 6. inner = inv(T2 - A22 T2 A22^T - Xt22)
  cpb(INA, 256, T2, 256, 256, 256, 1.f, 0.f);
  cpb(INA, 256, T2 + 128 * 256 + 128, 256, 128, 128, -1.f, 1.f);
  axpy(INA, XT22, 256 * 256, -1.f);
  inv(INV2, INA, 256);
  // 7. cross = [A12 @ T2[:,128:], BQ]
  gemm(CR, A12, T2 + 128, 256, 128, 256, 256, 256, 256, 0, 0, 1.f, 0.f);
  cpb(CR + 128, 256, BQ, 128, 256, 128, 1.f, 0.f);
  // 8. M1 = A12 T2 A12^T + Xt11 + cross inner cross^T + H1^T H1 + eps I
  gemm(TMPA, A12, T2, 256, 256, 256, 256, 256, 256, 0, 0, 1.f, 0.f);
  gemm(M1, TMPA, A12, 256, 256, 256, 256, 256, 256, 0, 1, 1.f, 0.f);
  axpy(M1, XT11, 256 * 256, 1.f);
  gemm(TMPA, CR, INV2, 256, 256, 256, 256, 256, 256, 0, 0, 1.f, 0.f);
  gemm(M1, TMPA, CR, 256, 256, 256, 256, 256, 256, 0, 1, 1.f, 1.f);
  gemm(M1, H1, H1, 256, 256, 256, 256, 256, 256, 1, 0, 1.f, 1.f);
  k_add_eye<<<1, 256, 0, stream>>>(M1, 256, 256, EPSV);
  // 9. T1 = M1 + A11 M1 A11^T (BR += TL)  [A11^2 == 0]
  cpb(M1 + 128 * 256 + 128, 256, M1, 256, 128, 128, 1.f, 1.f);
  // 10. P = blkdiag(inv(T1), inv(T2))
  inv(TMPB, M1, 256);
  inv(TMPA, T2, 256);
  fill(P, 512 * 512, 0.f);
  cpb(P, 512, TMPB, 256, 256, 256, 1.f, 0.f);
  cpb(P + 256 * 512 + 256, 512, TMPA, 256, 256, 256, 1.f, 0.f);
  // 11. A = [[A11, A12],[0, A22]] ; Bm = [B1; B2]
  fill(AM, 512 * 512, 0.f);
  k_set_diag<<<1, 256, 0, stream>>>(AM + 128 * 512, 512, (const float*)nullptr, 1.f, 128);
  cpb(AM + 256, 512, A12, 256, 256, 256, 1.f, 0.f);
  k_set_diag<<<1, 256, 0, stream>>>(AM + 256 * 512 + 384, 512, (const float*)nullptr, 1.f, 128);
  fill(BM, 512 * 128, 0.f);
  cpb(BM, 128, B1, 128, 256, 128, 1.f, 0.f);
  k_set_diag<<<1, 256, 0, stream>>>(BM + 384 * 128, 128, (const float*)nullptr, 1.f, 128);
  // 12. PA, PB
  gemm(TMPA, P, AM, 512, 512, 512, 512, 512, 512, 0, 0, 1.f, 0.f);
  gemm(PB, P, BM, 512, 128, 512, 512, 128, 128, 0, 0, 1.f, 0.f);
  // 13. F assembly (640x640)
  cpb(F, 640, P, 512, 512, 512, 1.f, 0.f);
  gemm(F, AM, TMPA, 512, 512, 512, 512, 512, 640, 1, 0, -1.f, 1.f);          // P - A^T P A
  gemm(F + 512, AM, PB, 512, 128, 512, 512, 128, 640, 1, 0, -1.f, 0.f);      // Fur
  gemm(F + 512 * 640, PB, AM, 128, 512, 512, 128, 512, 640, 1, 0, -1.f, 0.f);// Fur^T
  cpb(F + 512 * 640 + 512, 640, Q, 128, 128, 128, 1.f, 0.f);
  gemm(F + 512 * 640 + 512, BM, PB, 128, 128, 512, 128, 128, 640, 1, 0, -1.f, 1.f);
  k_symmetrize<<<(640 * 640 + 255) / 256, 256, 0, stream>>>(F, 640, 640);
  // 14. L = chol(F)
  k_cholesky<<<1, 1024, 0, stream>>>(F, 640, 640);
  // 15. Z1t = solve(L11, [0; diag(q)])^T  (Y holds solve result, 256x128)
  fill(Y, 256 * 128, 0.f);
  k_set_diag<<<1, 256, 0, stream>>>(Y + 128 * 128, 128, qv, 0.f, 128);
  k_trsm<<<1, 128, 0, stream>>>(F, 640, Y, 128, 256, 128);
  // 16. G = Y^T Y ; Sym = I - G/(1+tr) ; Sf = chol(Sym)
  gemm(G, Y, Y, 128, 128, 256, 128, 128, 128, 1, 0, 1.f, 0.f);
  k_trace<<<1, 1, 0, stream>>>(TR, G, 128, 128);
  k_make_sym<<<(128 * 128 + 255) / 256, 256, 0, stream>>>(SYM, G, TR, 128);
  k_cholesky<<<1, 1024, 0, stream>>>(SYM, 128, 128);
  // 17. Cayley of weight[128:]: U0 = wt[128:256], V = wt[256:512]
  const float* U0 = wt + 128 * 128;
  const float* V  = wt + 256 * 128;
  gemm(AC, V, V, 128, 128, 256, 128, 128, 128, 1, 0, 1.f, 0.f);
  k_skewadd<<<(128 * 128 + 255) / 256, 256, 0, stream>>>(AC, U0, 128);
  cpb(ACT, 128, AC, 128, 128, 128, 1.f, 0.f);
  k_add_eye<<<1, 256, 0, stream>>>(ACT, 128, 128, 1.f);
  inv(IIPA, ACT, 128);
  cpb(IMA, 128, AC, 128, 128, 128, -1.f, 0.f);
  k_add_eye<<<1, 256, 0, stream>>>(IMA, 128, 128, 1.f);
  gemm(UCAL, IIPA, IMA, 128, 128, 128, 128, 128, 128, 0, 0, 1.f, 0.f);
  gemm(UCAL + 128 * 128, V, IIPA, 256, 128, 128, 128, 128, 128, 0, 0, -2.f, 0.f);
  // 18. Z2t = Sf @ Ucal^T
  gemm(Z2T, SYM, UCAL, 128, 384, 128, 128, 128, 384, 0, 1, 1.f, 0.f);
  // 19. C2D = (Z1t @ L21^T + Z2t @ L22^T) / q
  gemm(C2D, Y, F + 256 * 640, 128, 384, 256, 128, 640, 384, 1, 1, 1.f, 0.f);
  gemm(C2D, Z2T, F + 256 * 640 + 256, 128, 384, 384, 384, 640, 384, 0, 1, 1.f, 1.f);
  k_divrow<<<(128 * 384 + 255) / 256, 256, 0, stream>>>(C2D, qv, 128, 384);
  // 20. mat = [[A12, B1],[C2, D]]
  cpb(MAT, 384, A12, 256, 256, 256, 1.f, 0.f);
  cpb(MAT + 256, 384, B1, 128, 256, 128, 1.f, 0.f);
  cpb(MAT + 256 * 384, 384, C2D, 384, 128, 256, 1.f, 0.f);
  cpb(MAT + 256 * 384 + 256, 384, C2D + 256, 384, 128, 128, 1.f, 0.f);
  // 21. Pack weights to f16 [co][k] fragment layout
  k_pack<<<(COUT * KTOT + 255) / 256, 256, 0, stream>>>(KH, MAT);
  // 22. Implicit-GEMM WMMA conv (one block per output row)
  dim3 grid(1, HH, BB);
  k_conv_wmma<<<grid, 256, 0, stream>>>(out, X16, KH, bia);
  (void)in_sizes; (void)n_in; (void)out_size; (void)ws_size;
}
